// TripletLoss_14465449853657
// MI455X (gfx1250) — matrix operand order repaired and checked by reading, not compiled, not run on hardware
//
#include <hip/hip_runtime.h>

// ---------------------------------------------------------------------------
// TripletLoss on MI455X (gfx1250, wave32)
//   dist = sqrt(clip(||xi||^2 + ||xj||^2 - 2 xi.xj, EPS))
//   loss = mean over (same-label, i!=j) pairs of max(0, 1 + dist_ij - neg_dist_i)
// Gram matrix via bf16x3 WMMA (hi/lo split) at bf16 rate, ~fp32 accuracy.
// B tiles staged into LDS by the Tensor Data Mover (TDM), double-buffered,
// synchronized with s_wait_tensorcnt + workgroup barriers.
// ---------------------------------------------------------------------------

#define NB   8192   // batch
#define DF   512    // feature dim
#define NC   128    // classes
#define MARGIN 1.0f
#define EPSD   1e-11f

typedef __attribute__((ext_vector_type(16))) __bf16 bf16x16;
typedef __attribute__((ext_vector_type(8)))  __bf16 bf16x8;
typedef __attribute__((ext_vector_type(8)))  float  f32x8;
typedef __attribute__((ext_vector_type(4)))  unsigned int u32x4;
typedef __attribute__((ext_vector_type(8)))  int          i32x8;
typedef __attribute__((ext_vector_type(4)))  int          i32x4;

// Dynamic-LDS layout for k_gemm_loss (bytes):
//   [0      , 8192 )  B-tile hi, buffer 0   (128 rows x 32 K x bf16)
//   [8192   , 16384)  B-tile lo, buffer 0
//   [16384  , 24576)  B-tile hi, buffer 1
//   [24576  , 32768)  B-tile lo, buffer 1
//   [32768  , 33792)  reduction floats (256)
//   [33792  , 34816)  reduction counts (256)
#define LDS_BUF0   0u
#define LDS_BUF1   16384u
#define LDS_TILE   8192u
#define LDS_RED_S  32768u
#define LDS_RED_C  33792u
#define LDS_TOTAL  34816u

__device__ __forceinline__ unsigned short f32_to_bf16_rne(float x) {
    unsigned u = __float_as_uint(x);
    unsigned r = u + 0x7FFFu + ((u >> 16) & 1u);
    return (unsigned short)(r >> 16);
}
__device__ __forceinline__ float bf16_bits_to_f32(unsigned short h) {
    return __uint_as_float(((unsigned)h) << 16);
}

// Issue one TDM transfer: 128 x 32 bf16 tile (rows stride DF) -> LDS at ldsOff.
// D# packing per CDNA5 ISA ch.8:
//   group0: [1:0]=count=1, [63:32]=lds_addr, [120:64]=global_addr, [127:126]=type=2
//   group1: data_size=1(2B), tensor_dim0=512, tensor_dim1=8192,
//           tile_dim0=32, tile_dim1=128, tensor_dim0_stride=512
__device__ __forceinline__ void tdm_load_tile(const unsigned short* gsrc, unsigned ldsOff) {
    unsigned long long ga = (unsigned long long)(__SIZE_TYPE__)gsrc;
    u32x4 g0;
    g0[0] = 1u;                                   // count=1, user descriptor
    g0[1] = ldsOff;                               // lds_addr
    g0[2] = (unsigned)ga;                         // global_addr[31:0]
    g0[3] = (unsigned)((ga >> 32) & 0x01FFFFFFull) | 0x80000000u; // addr[56:32] | type=2
    i32x8 g1;
    g1[0] = 0x00010000;   // wg_mask=0, data_size=1 (2 bytes)
    g1[1] = 0x02000000;   // tensor_dim0 = 512  (bits 63:48)
    g1[2] = 0x20000000;   // tensor_dim1 = 8192 (bits 95:80); dim0 hi16 = 0
    g1[3] = 0x00200000;   // tile_dim0 = 32 (bits 127:112); dim1 hi16 = 0
    g1[4] = 128;          // tile_dim1 = 128
    g1[5] = 512;          // tensor_dim0_stride = 512
    g1[6] = 0;
    g1[7] = 0;
    i32x4 z4 = {0, 0, 0, 0};
    i32x8 z8 = {0, 0, 0, 0, 0, 0, 0, 0};
    // 6-arg form (clang-23 / therock-10.0 headers)
    __builtin_amdgcn_tensor_load_to_lds(g0, g1, z4, z4, z8, 0);
}

// ---------------- kernel 0: zero the class histogram -----------------------
__global__ void k_init(unsigned* counts) {
    if (threadIdx.x < NC) counts[threadIdx.x] = 0u;
}

// ---------------- kernel 1: bf16 hi/lo split + row sum-of-squares ----------
__global__ void k_prep(const float* __restrict__ f,
                       unsigned short* __restrict__ hi,
                       unsigned short* __restrict__ lo,
                       float* __restrict__ sq) {
    const int r = blockIdx.x;
    const int t = threadIdx.x;
    __shared__ float red[256];
    float s = 0.f;
    for (int d = t; d < DF; d += 256) {
        float x = f[(size_t)r * DF + d];
        s += x * x;
        unsigned short hb = f32_to_bf16_rne(x);
        float xl = x - bf16_bits_to_f32(hb);
        unsigned short lb = f32_to_bf16_rne(xl);
        hi[(size_t)r * DF + d] = hb;
        lo[(size_t)r * DF + d] = lb;
    }
    red[t] = s;
    __syncthreads();
    for (int o = 128; o > 0; o >>= 1) {
        if (t < o) red[t] += red[t + o];
        __syncthreads();
    }
    if (t == 0) sq[r] = red[0];
}

// ---------------- kernel 2: class histogram --------------------------------
__global__ void k_hist(const int* __restrict__ labels, unsigned* __restrict__ counts) {
    int g = blockIdx.x * blockDim.x + threadIdx.x;
    if (g < NB) atomicAdd(&counts[labels[g]], 1u);
}

// ---------------- kernel 3: pick k-th negative, exact fp32 neg_dist --------
__global__ void k_negpick(const float* __restrict__ f,
                          const int* __restrict__ labels,
                          const float* __restrict__ negu,
                          const unsigned* __restrict__ counts,
                          const float* __restrict__ sq,
                          float* __restrict__ negdist) {
    int g = blockIdx.x * blockDim.x + threadIdx.x;
    if (g >= NB) return;
    const int myl = labels[g];
    const int nneg = NB - (int)counts[myl];
    if (nneg <= 0) { negdist[g] = -1.0f; return; }  // sentinel: no valid negative
    int k = (int)floorf(negu[g] * (float)nneg);
    if (k < 0) k = 0;
    if (k > nneg - 1) k = nneg - 1;
    // k-th (0-based) column j in ascending order with labels[j] != myl
    int c = -1, fnd = 0;
    for (int j = 0; j < NB; ++j) {
        if (labels[j] != myl) {
            if (++c == k) { fnd = j; break; }
        }
    }
    float acc = 0.f;
    const float* xi = f + (size_t)g * DF;
    const float* xj = f + (size_t)fnd * DF;
    for (int d = 0; d < DF; ++d) acc = fmaf(xi[d], xj[d], acc);
    float d2 = sq[g] + sq[fnd] - 2.0f * acc;
    negdist[g] = sqrtf(fmaxf(d2, EPSD));
}

// ---------------- kernel 4: WMMA GEMM (bf16x3) fused with loss -------------
// Block: 256 threads = 8 waves. Tile: 128x128 of the Gram matrix.
// Wave w owns rows [row0, row0+16) x all 128 cols -> 8 WMMA subtiles.
// B tile (128x32, shared by all 8 waves) staged into LDS by TDM, double-buffered.
__global__ void __launch_bounds__(256)
k_gemm_loss(const unsigned short* __restrict__ hi,
            const unsigned short* __restrict__ lo,
            const int* __restrict__ labels,
            const float* __restrict__ sq,
            const float* __restrict__ negdist,
            float* __restrict__ psum,
            unsigned* __restrict__ pcnt) {
    extern __shared__ __align__(32) char smem[];

    const int tid  = threadIdx.x;
    const int wave = tid >> 5;
    const int lane = tid & 31;
    const int m    = lane & 15;   // row-in-tile (A) / col-in-subtile (B,C)
    const int kh   = lane >> 4;   // K half selector per ISA 16-bit layouts

    const int row0    = blockIdx.y * 128 + wave * 16;
    const int colBase = blockIdx.x * 128;
    const int arow    = row0 + m;

    f32x8 acc[8];
#pragma unroll
    for (int t = 0; t < 8; ++t) acc[t] = (f32x8)(0.f);

    const unsigned short* aHi = hi + (size_t)arow * DF;
    const unsigned short* aLo = lo + (size_t)arow * DF;
    const unsigned short* bHiBase = hi + (size_t)colBase * DF;
    const unsigned short* bLoBase = lo + (size_t)colBase * DF;

    // Prologue: wave0 kicks TDM for K-step 0 into buffer 0 (hi + lo tiles).
    if (wave == 0) {
        tdm_load_tile(bHiBase, LDS_BUF0);
        tdm_load_tile(bLoBase, LDS_BUF0 + LDS_TILE);
    }

    const int nsteps = DF / 32;   // 16
    for (int i = 0; i < nsteps; ++i) {
        const int k0 = i * 32;
        const unsigned curOff = (i & 1) ? LDS_BUF1 : LDS_BUF0;

        if (wave == 0) {
            if (i + 1 < nsteps) {
                const unsigned nxtOff = ((i + 1) & 1) ? LDS_BUF1 : LDS_BUF0;
                tdm_load_tile(bHiBase + (k0 + 32), nxtOff);
                tdm_load_tile(bLoBase + (k0 + 32), nxtOff + LDS_TILE);
                __builtin_amdgcn_s_wait_tensorcnt(2); // current buffer complete
            } else {
                __builtin_amdgcn_s_wait_tensorcnt(0);
            }
        }
        __syncthreads();  // publish current B buffer to all waves

        // A fragment 16x32 (per-wave, from global/L2):
        // lanes 0-15 hold K[k0+0..15], lanes 16-31 hold K[k0+16..31]
        bf16x8 a0h = *reinterpret_cast<const bf16x8*>(aHi + k0 + 16 * kh);
        bf16x8 a1h = *reinterpret_cast<const bf16x8*>(aHi + k0 + 8 + 16 * kh);
        bf16x8 a0l = *reinterpret_cast<const bf16x8*>(aLo + k0 + 16 * kh);
        bf16x8 a1l = *reinterpret_cast<const bf16x8*>(aLo + k0 + 8 + 16 * kh);
        bf16x16 ah, al;
#pragma unroll
        for (int q = 0; q < 8; ++q) {
            ah[q] = a0h[q]; ah[8 + q] = a1h[q];
            al[q] = a0l[q]; al[8 + q] = a1l[q];
        }

        const char* bb = smem + curOff;
#pragma unroll
        for (int t = 0; t < 8; ++t) {
            // B fragment 32x16 of X^T: lane = 16 consecutive bf16 of LDS row n
            const int nl = t * 16 + m;                    // local column 0..127
            bf16x16 bh = *reinterpret_cast<const bf16x16*>(bb + nl * 64 + kh * 32);
            bf16x16 bl = *reinterpret_cast<const bf16x16*>(bb + LDS_TILE + nl * 64 + kh * 32);
            // fp32-accurate product: hi*hi + hi*lo + lo*hi (lo*lo ~ 2^-32, dropped)
            acc[t] = __builtin_amdgcn_wmma_f32_16x16x32_bf16(
                false, ah, false, bh, (short)0, acc[t], false, false);
            acc[t] = __builtin_amdgcn_wmma_f32_16x16x32_bf16(
                false, ah, false, bl, (short)0, acc[t], false, false);
            acc[t] = __builtin_amdgcn_wmma_f32_16x16x32_bf16(
                false, al, false, bh, (short)0, acc[t], false, false);
        }
        __syncthreads();  // all waves done reading before buffer is re-filled
    }

    // Epilogue: C layout (32-bit, 16x16): VGPR v, lanes 0-15 -> M=v, lanes 16-31 -> M=v+8
    int   li[8];
    float sqi[8], ndi[8];
#pragma unroll
    for (int v = 0; v < 8; ++v) {
        const int i = row0 + v + 8 * kh;
        li[v]  = labels[i];
        sqi[v] = sq[i];
        ndi[v] = negdist[i];
    }

    float    s = 0.f;
    unsigned c = 0u;
#pragma unroll
    for (int t = 0; t < 8; ++t) {
        const int j  = colBase + t * 16 + m;
        const int lj = labels[j];
        const float sqj = sq[j];
#pragma unroll
        for (int v = 0; v < 8; ++v) {
            const int i = row0 + v + 8 * kh;
            if (li[v] == lj && i != j && ndi[v] >= 0.f) {
                float d2   = sqi[v] + sqj - 2.0f * acc[t][v];
                float dist = sqrtf(fmaxf(d2, EPSD));
                s += fmaxf(MARGIN + dist - ndi[v], 0.f);
                c += 1u;
            }
        }
    }

    float*    ss = (float*)(smem + LDS_RED_S);
    unsigned* sc = (unsigned*)(smem + LDS_RED_C);
    ss[tid] = s; sc[tid] = c;
    __syncthreads();
    for (int o = 128; o > 0; o >>= 1) {
        if (tid < o) { ss[tid] += ss[tid + o]; sc[tid] += sc[tid + o]; }
        __syncthreads();
    }
    if (tid == 0) {
        const int bid = blockIdx.y * gridDim.x + blockIdx.x;
        psum[bid] = ss[0];
        pcnt[bid] = sc[0];
    }
}

// ---------------- kernel 5: deterministic final reduce ---------------------
__global__ void k_final(const float* __restrict__ psum,
                        const unsigned* __restrict__ pcnt,
                        float* __restrict__ out, int nblk) {
    const int t = threadIdx.x;
    __shared__ float    ss[256];
    __shared__ unsigned sc[256];
    float    s = 0.f;
    unsigned c = 0u;
    for (int g = t; g < nblk; g += 256) { s += psum[g]; c += pcnt[g]; }
    ss[t] = s; sc[t] = c;
    __syncthreads();
    for (int o = 128; o > 0; o >>= 1) {
        if (t < o) { ss[t] += ss[t + o]; sc[t] += sc[t + o]; }
        __syncthreads();
    }
    if (t == 0) out[0] = (sc[0] > 0u) ? (ss[0] / (float)sc[0]) : 0.f;
}

// ---------------------------------------------------------------------------
extern "C" void kernel_launch(void* const* d_in, const int* in_sizes, int n_in,
                              void* d_out, int out_size, void* d_ws, size_t ws_size,
                              hipStream_t stream) {
    (void)in_sizes; (void)n_in; (void)out_size; (void)ws_size;
    const float* features = (const float*)d_in[0];
    const int*   labels   = (const int*)d_in[1];   // jnp int64 canonicalized -> int32
    const float* negu     = (const float*)d_in[2];
    float*       out      = (float*)d_out;

    // Workspace layout (all offsets 256-aligned where it matters)
    char* ws = (char*)d_ws;
    unsigned*       counts  = (unsigned*)(ws + 0);                 // 128 u32
    float*          sq      = (float*)(ws + 512);                  // 8192 f32
    float*          negdist = (float*)(ws + 512 + 32768);          // 8192 f32
    unsigned short* hi      = (unsigned short*)(ws + 66048);       // 8192*512 bf16 (8 MB)
    unsigned short* lo      = (unsigned short*)(ws + 66048 + 8388608);
    float*          psum    = (float*)(ws + 16843264);             // 4096 f32
    unsigned*       pcnt    = (unsigned*)(ws + 16843264 + 16384);  // 4096 u32

    k_init   <<<1, 128, 0, stream>>>(counts);
    k_prep   <<<NB, 256, 0, stream>>>(features, hi, lo, sq);
    k_hist   <<<NB / 256, 256, 0, stream>>>(labels, counts);
    k_negpick<<<NB / 256, 256, 0, stream>>>(features, labels, negu, counts, sq, negdist);
    k_gemm_loss<<<dim3(64, 64), 256, LDS_TOTAL, stream>>>(hi, lo, labels, sq, negdist, psum, pcnt);
    k_final  <<<1, 256, 0, stream>>>(psum, pcnt, out, 64 * 64);
}
